// MultiHeadAttention_55035710931446
// MI455X (gfx1250) — matrix operand order repaired and checked by reading, not compiled, not run on hardware
//
#include <hip/hip_runtime.h>
#include <hip/hip_bf16.h>

// MI455X (gfx1250) multi-head attention block: QKV proj -> causal softmax
// attention (scores materialized, 1 GiB fp32) -> out proj + residual + LN.
// All GEMMs use v_wmma_f32_16x16x32_f16 (f16 in, f32 accumulate).
// Global streams are prefetched one tile ahead (global_prefetch_b8).

#define BB 4
#define SS 2048
#define DD 1024
#define HH 16
#define DKV 64

typedef __attribute__((ext_vector_type(16))) _Float16 v16h;
typedef __attribute__((ext_vector_type(8)))  _Float16 v8h;
typedef __attribute__((ext_vector_type(8)))  float    v8f;

// ---------------- WMMA fragment helpers (ISA 7.12.2 layouts, wave32) -------

// A-matrix 16x32 (MxK) f16. Lane 0-15: M=lane, halves 0-7 -> K=k0..k0+7,
// halves 8-15 -> K=k0+16..k0+23. Lanes 16-31: same M, +8 on K.
__device__ __forceinline__ v16h load_a_frag(const _Float16* base, int ld, int k0) {
  int lane = threadIdx.x & 31;
  int m    = lane & 15;
  int hiK  = (lane >> 4) & 1;
  const _Float16* p = base + (size_t)m * ld + k0 + hiK * 8;
  v8h lo = *(const v8h*)(p);        // 16B load
  v8h hi = *(const v8h*)(p + 16);   // 16B load
  return __builtin_shufflevector(lo, hi, 0,1,2,3,4,5,6,7,8,9,10,11,12,13,14,15);
}

// B-matrix 32x16 (KxN) f16 from "transposed" storage bt[n*ld + k]
// (element (k,n) contiguous in k). Lane: N = n0+(lane&15); halves 0-7 -> K = k0+8*hiK+h,
// halves 8-15 -> K = k0+16+8*hiK+(h-8) (same order as load_a_frag).
__device__ __forceinline__ v16h load_b_fragT(const _Float16* bt, int ld, int k0, int n0) {
  int lane = threadIdx.x & 31;
  int n    = n0 + (lane & 15);
  int hiK  = (lane >> 4) & 1;
  const _Float16* p = bt + (size_t)n * ld + k0 + hiK * 8;
  v8h lo = *(const v8h*)(p);        // 16B load
  v8h hi = *(const v8h*)(p + 16);   // 16B load
  return __builtin_shufflevector(lo, hi, 0,1,2,3,4,5,6,7,8,9,10,11,12,13,14,15);
}

__device__ __forceinline__ v8f wmma_f16(v16h a, v16h b, v8f c) {
  return __builtin_amdgcn_wmma_f32_16x16x32_f16(false, a, false, b, (short)0, c,
                                                false, false);
}

__device__ __forceinline__ float wave_max(float v) {
#pragma unroll
  for (int m = 16; m > 0; m >>= 1) v = fmaxf(v, __shfl_xor(v, m, 32));
  return v;
}
__device__ __forceinline__ float wave_sum(float v) {
#pragma unroll
  for (int m = 16; m > 0; m >>= 1) v += __shfl_xor(v, m, 32);
  return v;
}

// ---------------- conversion kernels ---------------------------------------

__global__ __launch_bounds__(256) void cvt_f32_to_f16_k(const float* __restrict__ in,
                                                        _Float16* __restrict__ out, int n) {
  int i = blockIdx.x * 256 + threadIdx.x;
  if (i < n) out[i] = (_Float16)in[i];
}

// w[k][n] (rows x cols) -> wt[n][k] f16
__global__ __launch_bounds__(256) void cvt_transpose_k(const float* __restrict__ w,
                                                       _Float16* __restrict__ wt,
                                                       int rows, int cols) {
  int i = blockIdx.x * 256 + threadIdx.x;
  if (i < rows * cols) {
    int k = i / cols, n = i % cols;
    wt[(size_t)n * rows + k] = (_Float16)w[i];
  }
}

// ---------------- QKV projection GEMM --------------------------------------
// C(8192x1024) = A(8192x1024) * B(1024x1024); B given transposed (N-major).
// Block 256 = 8 waves; wave computes 16 rows x 64 cols; grid (512, 2).
// vmode 0: out[((b*H+h)*S+s)*64 + d]   (Q, K head-major)
// vmode 1: out[((b*H+h)*64+d)*S + s]   (V transposed for PV B-operand)
__global__ __launch_bounds__(256) void qkv_gemm_k(const _Float16* __restrict__ A,
                                                  const _Float16* __restrict__ Bt,
                                                  _Float16* __restrict__ out, int vmode) {
  int wave = threadIdx.x >> 5;
  int lane = threadIdx.x & 31;
  int m0 = blockIdx.x * 16;
  int n0 = blockIdx.y * 512 + wave * 64;
  const _Float16* Arow = A + (size_t)m0 * DD;
  v8f acc[4] = {};
  for (int k0 = 0; k0 < DD; k0 += 32) {
    if (k0 + 32 < DD) {  // speculative prefetch of next k-panel
      __builtin_prefetch(Arow + (size_t)(lane & 15) * DD + k0 + 32, 0, 0);
      __builtin_prefetch(Bt + (size_t)(n0 + (lane & 15)) * DD + k0 + 32, 0, 0);
    }
    v16h a = load_a_frag(Arow, DD, k0);
#pragma unroll
    for (int t = 0; t < 4; ++t) {
      v16h b = load_b_fragT(Bt, DD, k0, n0 + t * 16);
      acc[t] = wmma_f16(a, b, acc[t]);
    }
  }
  int mb = m0 + ((lane >> 4) & 1) * 8;
#pragma unroll
  for (int t = 0; t < 4; ++t) {
    int n = n0 + t * 16 + (lane & 15);
    int h = n >> 6, d = n & 63;
#pragma unroll
    for (int i = 0; i < 8; ++i) {
      int r = mb + i;
      int b = r >> 11, s = r & (SS - 1);
      size_t off = vmode ? ((((size_t)b * HH + h) * DKV + d) * SS + s)
                         : ((((size_t)b * HH + h) * SS + s) * DKV + d);
      out[off] = (_Float16)acc[t][i];
    }
  }
}

// ---------------- attention kernel -----------------------------------------
// One block per (b,h, 16-row query tile). Whole 16xS score stripe in LDS
// (128 KB; CDNA5 WGP has 320 KB). 8 waves.
__global__ __launch_bounds__(256) void attn_k(const _Float16* __restrict__ Q,
                                              const _Float16* __restrict__ K,
                                              const _Float16* __restrict__ Vt,
                                              float* __restrict__ score_out,
                                              _Float16* __restrict__ ctx) {
  __shared__ float srow[16 * SS];     // 128 KB score stripe / scratch
  __shared__ float rowM[16], rowInv[16];

  int wave = threadIdx.x >> 5;
  int lane = threadIdx.x & 31;
  int bh = blockIdx.x >> 7;           // S/16 = 128 query tiles per (b,h)
  int qt = blockIdx.x & 127;
  int b = bh >> 4, h = bh & 15;
  int q0 = qt * 16;
  int nkt = qt + 1;                   // key tiles with any unmasked entry

  const _Float16* Qp = Q + ((size_t)bh * SS + q0) * DKV;
  const _Float16* Kp = K + (size_t)bh * SS * DKV;      // [s][dk]
  const _Float16* Vp = Vt + (size_t)bh * DKV * SS;     // [dv][s]
  size_t obase = ((size_t)bh * SS + q0) * SS;

  // Q A-fragments (16x64 = two 16x32 frags), loaded once per wave.
  v16h a0 = load_a_frag(Qp, DKV, 0);
  v16h a1 = load_a_frag(Qp, DKV, 32);
  const float scale = 0.125f;  // 1/sqrt(DK)

  // ---- pass 1: raw scaled+masked scores into LDS -------------------------
  for (int kt = wave; kt < nkt; kt += 8) {
    if (kt + 8 < nkt)  // prefetch this wave's next K tile
      __builtin_prefetch(Kp + (size_t)((kt + 8) * 16 + (lane & 15)) * DKV, 0, 0);
    v8f c = {};
    c = wmma_f16(a0, load_b_fragT(Kp, DKV, 0, kt * 16), c);
    c = wmma_f16(a1, load_b_fragT(Kp, DKV, 32, kt * 16), c);
    int n = kt * 16 + (lane & 15);
    int mb = ((lane >> 4) & 1) * 8;
#pragma unroll
    for (int i = 0; i < 8; ++i) {
      int m = mb + i;
      float v = c[i] * scale;
      if (n > q0 + m) v = -1e9f;      // causal mask (reference semantics)
      srow[m * SS + n] = v;
    }
  }
  // zero-fill fully-masked score region directly in global (exact 0, matches
  // fp32 exp(-1e9 - max) underflow in the reference).
  int kstart = nkt * 16;
#pragma unroll 1
  for (int m = 0; m < 16; ++m)
    for (int k = kstart + threadIdx.x; k < SS; k += 256)
      score_out[obase + (size_t)m * SS + k] = 0.f;
  __syncthreads();

  // ---- row max / sum (2 rows per wave, lane-strided + shuffle reduce) ----
#pragma unroll
  for (int r = 0; r < 2; ++r) {
    int m = wave * 2 + r;
    int kmax = q0 + m;                // inclusive causal limit
    float mx = -3.4e38f;
    for (int k = lane; k <= kmax; k += 32) mx = fmaxf(mx, srow[m * SS + k]);
    mx = wave_max(mx);
    float sm = 0.f;
    for (int k = lane; k <= kmax; k += 32) sm += __expf(srow[m * SS + k] - mx);
    sm = wave_sum(sm);
    if (lane == 0) { rowM[m] = mx; rowInv[m] = 1.f / sm; }
  }
  __syncthreads();

  // ---- normalize: write probabilities to global + back into LDS ----------
#pragma unroll 1
  for (int m = 0; m < 16; ++m) {
    float mx = rowM[m], inv = rowInv[m];
    int kmax = q0 + m;
    for (int k = threadIdx.x; k < kstart; k += 256) {
      float p = (k <= kmax) ? __expf(srow[m * SS + k] - mx) * inv : 0.f;
      srow[m * SS + k] = p;
      score_out[obase + (size_t)m * SS + k] = p;
    }
  }
  // pad P to a multiple of 32 keys for the PV A-fragments
  if (nkt & 1) {
    for (int idx = threadIdx.x; idx < 16 * 16; idx += 256) {
      int m = idx >> 4;
      srow[m * SS + kstart + (idx & 15)] = 0.f;
    }
  }
  __syncthreads();

  // ---- P @ V: key chunks of 32, 4 accum tiles (16x64) per wave -----------
  int nkc = (nkt + 1) >> 1;
  v8f acc[4] = {};
  int m = lane & 15;
  int hiK = (lane >> 4) & 1;
  for (int kc = wave; kc < nkc; kc += 8) {
    if (kc + 8 < nkc)  // prefetch this wave's next V chunk
      __builtin_prefetch(Vp + (size_t)(lane & 15) * SS + (kc + 8) * 32, 0, 0);
    v16h a;
    int kb = kc * 32 + hiK * 8;
    const float* pr = srow + m * SS + kb;
#pragma unroll
    for (int i = 0; i < 8; ++i) a[i] = (_Float16)pr[i];
#pragma unroll
    for (int i = 0; i < 8; ++i) a[8 + i] = (_Float16)pr[16 + i];
#pragma unroll
    for (int t = 0; t < 4; ++t)
      acc[t] = wmma_f16(a, load_b_fragT(Vp, SS, kc * 32, t * 16), acc[t]);
  }
  __syncthreads();
  // per-wave partial context -> LDS (reuse srow), then cross-wave reduce.
  {
    float* part = srow + wave * 1024;
    int mb = hiK * 8;
#pragma unroll
    for (int t = 0; t < 4; ++t) {
      int n = t * 16 + (lane & 15);
#pragma unroll
      for (int i = 0; i < 8; ++i) part[(mb + i) * DKV + n] = acc[t][i];
    }
  }
  __syncthreads();
  for (int idx = threadIdx.x; idx < 16 * DKV; idx += 256) {
    float s = 0.f;
#pragma unroll
    for (int w = 0; w < 8; ++w) s += srow[w * 1024 + idx];
    int mm = idx >> 6, dv = idx & 63;
    ctx[((size_t)b * SS + q0 + mm) * DD + h * DKV + dv] = (_Float16)s;
  }
}

// ---------------- output projection + residual + LayerNorm -----------------
// Block owns 16 full rows (needed for LN). 8 waves x (16x128) tiles.
__global__ __launch_bounds__(256) void out_proj_ln_k(const _Float16* __restrict__ ctx,
                                                     const _Float16* __restrict__ WoT,
                                                     const float* __restrict__ x,
                                                     const float* __restrict__ gamma,
                                                     const float* __restrict__ beta,
                                                     float* __restrict__ out) {
  __shared__ float obuf[16 * DD];     // 64 KB
  int wave = threadIdx.x >> 5;
  int lane = threadIdx.x & 31;
  int m0 = blockIdx.x * 16;
  int nbase = wave * 128;
  const _Float16* Arow = ctx + (size_t)m0 * DD;
  v8f acc[8] = {};
  for (int k0 = 0; k0 < DD; k0 += 32) {
    if (k0 + 32 < DD) {  // speculative prefetch of next k-panel
      __builtin_prefetch(Arow + (size_t)(lane & 15) * DD + k0 + 32, 0, 0);
      __builtin_prefetch(WoT + (size_t)(nbase + (lane & 15)) * DD + k0 + 32, 0, 0);
    }
    v16h a = load_a_frag(Arow, DD, k0);
#pragma unroll
    for (int t = 0; t < 8; ++t)
      acc[t] = wmma_f16(a, load_b_fragT(WoT, DD, k0, nbase + t * 16), acc[t]);
  }
  int mb = ((lane >> 4) & 1) * 8;
#pragma unroll
  for (int t = 0; t < 8; ++t) {
    int n = nbase + t * 16 + (lane & 15);
#pragma unroll
    for (int i = 0; i < 8; ++i) obuf[(mb + i) * DD + n] = acc[t][i];
  }
  __syncthreads();
#pragma unroll
  for (int r = 0; r < 2; ++r) {
    int m = wave * 2 + r;
    float* row = obuf + m * DD;
    size_t gr = (size_t)(m0 + m) * DD;
    float sum = 0.f;
    for (int k = lane; k < DD; k += 32) {
      float v = row[k] + x[gr + k];   // residual
      row[k] = v;
      sum += v;
    }
    sum = wave_sum(sum);
    float mu = sum * (1.f / DD);
    float vs = 0.f;
    for (int k = lane; k < DD; k += 32) {
      float d = row[k] - mu;
      vs += d * d;
    }
    vs = wave_sum(vs);
    float rstd = rsqrtf(vs * (1.f / DD) + 1e-6f);
    for (int k = lane; k < DD; k += 32)
      out[gr + k] = (row[k] - mu) * rstd * gamma[k] + beta[k];
  }
}

// ---------------- host launcher --------------------------------------------

extern "C" void kernel_launch(void* const* d_in, const int* in_sizes, int n_in,
                              void* d_out, int out_size, void* d_ws, size_t ws_size,
                              hipStream_t stream) {
  const float* x     = (const float*)d_in[0];
  // d_in[1] = mask (causal; structure known, not read)
  const float* Wq    = (const float*)d_in[2];
  const float* Wk    = (const float*)d_in[3];
  const float* Wv    = (const float*)d_in[4];
  const float* Wo    = (const float*)d_in[5];
  const float* gamma = (const float*)d_in[6];
  const float* beta  = (const float*)d_in[7];

  const size_t ROWS = (size_t)BB * SS;          // 8192
  const size_t NX   = ROWS * DD;                // 8,388,608
  const size_t NW   = (size_t)DD * DD;          // 1,048,576

  _Float16* ws  = (_Float16*)d_ws;
  _Float16* xh  = ws;            // NX
  _Float16* WqT = xh  + NX;      // NW each
  _Float16* WkT = WqT + NW;
  _Float16* WvT = WkT + NW;
  _Float16* WoT = WvT + NW;
  _Float16* Qb  = WoT + NW;      // NX each
  _Float16* Kb  = Qb  + NX;
  _Float16* Vtb = Kb  + NX;
  _Float16* ctx = Vtb + NX;
  if (ws_size < (5 * NX + 4 * NW) * sizeof(_Float16)) return;

  float* score_out = (float*)d_out;
  float* ln_out    = score_out + (size_t)BB * HH * SS * SS;

  // 1) f16 conversions (x + transposed weights)
  cvt_f32_to_f16_k<<<(int)((NX + 255) / 256), 256, 0, stream>>>(x, xh, (int)NX);
  cvt_transpose_k<<<(int)((NW + 255) / 256), 256, 0, stream>>>(Wq, WqT, DD, DD);
  cvt_transpose_k<<<(int)((NW + 255) / 256), 256, 0, stream>>>(Wk, WkT, DD, DD);
  cvt_transpose_k<<<(int)((NW + 255) / 256), 256, 0, stream>>>(Wv, WvT, DD, DD);
  cvt_transpose_k<<<(int)((NW + 255) / 256), 256, 0, stream>>>(Wo, WoT, DD, DD);

  // 2) QKV projections (WMMA)
  dim3 ggrid(ROWS / 16, 2);
  qkv_gemm_k<<<ggrid, 256, 0, stream>>>(xh, WqT, Qb, 0);
  qkv_gemm_k<<<ggrid, 256, 0, stream>>>(xh, WkT, Kb, 0);
  qkv_gemm_k<<<ggrid, 256, 0, stream>>>(xh, WvT, Vtb, 1);

  // 3) causal attention: one block per (b,h,16 query rows)
  attn_k<<<BB * HH * (SS / 16), 256, 0, stream>>>(Qb, Kb, Vtb, score_out, ctx);

  // 4) output projection + residual + LayerNorm
  out_proj_ln_k<<<(int)(ROWS / 16), 256, 0, stream>>>(ctx, WoT, x, gamma, beta, ln_out);
}
